// SlowFast_77970836291601
// MI455X (gfx1250) — matrix operand order repaired and checked by reading, not compiled, not run on hardware
//
#include <hip/hip_runtime.h>
#include <cstdint>

typedef unsigned short u16;
typedef unsigned int   u32;

typedef __attribute__((ext_vector_type(16))) __bf16 bfvec16;
typedef __attribute__((ext_vector_type(8)))  float  fvec8;
typedef __attribute__((ext_vector_type(4)))  u32    v4u;
typedef __attribute__((ext_vector_type(8)))  int    v8i;
typedef __attribute__((ext_vector_type(4)))  int    v4i;

#if __has_builtin(__builtin_amdgcn_tensor_load_to_lds) && __has_builtin(__builtin_amdgcn_s_wait_tensorcnt)
#define USE_TDM 1
#else
#define USE_TDM 0
#endif

struct alignas(16) U16B { u32 d[4]; };

union FragAB { bfvec16 v; u16 u[16]; U16B q[2]; };
union FragC  { fvec8 v; float f[8]; };

// ---------------- constants ----------------
constexpr int NB = 4, CC = 256, TT = 400, VV = 79, VP = 80, SS = 2, NI = 64;
constexpr int LQ = TT * VP;                       // 32000 columns per (t,v) plane
constexpr size_t PLANE = (size_t)CC * LQ;         // 8,192,000 elems per batch plane

// ---------------- helpers ----------------
__device__ __forceinline__ u16 f2bf(float f) {
  u32 u = __float_as_uint(f);
  u += 0x7FFFu + ((u >> 16) & 1u);                // round-to-nearest-even
  return (u16)(u >> 16);
}
__device__ __forceinline__ float bf2f(u16 h) { return __uint_as_float(((u32)h) << 16); }
__device__ __forceinline__ float lrelu(float x) { return x >= 0.f ? x : 0.1f * x; }

__device__ __forceinline__ fvec8 wmma_bf16(bfvec16 a, bfvec16 b, fvec8 c) {
  return __builtin_amdgcn_wmma_f32_16x16x32_bf16(false, a, false, b, (short)0, c, false, false);
}

// A fragment (16x32 bf16, row-major A[M][lda]); lanes 0-15: K 0-15; lanes 16-31: K 16-31
__device__ __forceinline__ void load_a(FragAB& f, const u16* A, int row, int lda, int k0, int lane) {
  const int h = lane >> 4;
  const u16* p = A + (size_t)row * lda + k0 + h * 16;
  f.q[0] = *reinterpret_cast<const U16B*>(p);
  f.q[1] = *reinterpret_cast<const U16B*>(p + 8);
}

// ---------------- prep kernels ----------------
__global__ void k_prep_x(const float* __restrict__ x, u16* __restrict__ xbf) {
  size_t idx = (size_t)blockIdx.x * 256 + threadIdx.x;   // over N*C*T*VP
  int v = (int)(idx % VP);
  size_t r = idx / VP;                                   // ((n*C+c)*T+t)
  u16 o = 0;
  if (v < VV) o = f2bf(x[r * VV + v]);
  xbf[idx] = o;
}

__global__ void k_prep_w(const float* __restrict__ win, const float* __restrict__ wout,
                         const float* __restrict__ wff, const float* __restrict__ wt,
                         u16* __restrict__ winb, u16* __restrict__ woutb,
                         u16* __restrict__ wffb, u16* __restrict__ wtb) {
  int idx = blockIdx.x * 256 + threadIdx.x;
  if (idx < 65536) { winb[idx] = f2bf(win[idx]); return; }
  idx -= 65536;
  if (idx < 131072) { woutb[idx] = f2bf(wout[idx]); return; }
  idx -= 131072;
  if (idx < 65536) { wffb[idx] = f2bf(wff[idx]); return; }
  idx -= 65536;                                          // w_t: [o][dt*256+c] <- wt[o][c][dt]
  int o = idx / 768, rr = idx % 768, dt = rr / 256, c = rr % 256;
  wtb[idx] = f2bf(wt[((size_t)(o * 256 + c)) * 3 + dt]);
}

__global__ void k_prep_bias2(const float* __restrict__ win, const float* __restrict__ bin,
                             float* __restrict__ bias2) {
  int idx = blockIdx.x * 256 + threadIdx.x;              // 256*80
  int o = idx / VP, v = idx % VP;
  float acc = 0.f;
  if (v < VV) {
    const float kln = logf(10000.f) / (float)CC;
    for (int c = 0; c < CC; ++c) {
      float d = __expf(-(float)(c & ~1) * kln);
      float ang = (float)v * d;
      float pe = (c & 1) ? __cosf(ang) : __sinf(ang);
      acc += win[o * CC + c] * pe;
    }
    acc += bin[o];
  }
  bias2[idx] = acc;
}

__global__ void k_prep_bn(const float* b_out, const float* g1, const float* bt1, const float* m1, const float* v1,
                          const float* b_ff, const float* g2, const float* bt2, const float* m2, const float* v2,
                          const float* b_t,  const float* g3, const float* bt3, const float* m3, const float* v3,
                          float* __restrict__ bnp) {
  int i = threadIdx.x;                                   // 256
  float sD = g1[i] * rsqrtf(v1[i] + 1e-5f);
  bnp[i]         = sD;
  bnp[256 + i]   = (b_out[i] - m1[i]) * sD + bt1[i];
  float sE = g2[i] * rsqrtf(v2[i] + 1e-5f);
  bnp[512 + i]   = sE;
  bnp[768 + i]   = (b_ff[i] - m2[i]) * sE + bt2[i];
  float sF = g3[i] * rsqrtf(v3[i] + 1e-5f);
  bnp[1024 + i]  = sF;
  bnp[1280 + i]  = (b_t[i] - m3[i]) * sF + bt3[i];
}

__global__ void k_zero_att(float* __restrict__ a) {
  int idx = blockIdx.x * 256 + threadIdx.x;              // 8*6400
  a[idx] = 0.f;
}

// ---------------- main tiled GEMM (stages A, D, E) ----------------
// MODE 0: out = bf16(acc + bias2[o][v])                       (qk projection)
// MODE 1: out = bf16(lrelu(x + acc*scale[o] + shift[o]))      (W_out / W_ff + BN + residual)
template <int MODE>
__global__ __launch_bounds__(256) void k_gemm_main(
    const u16* __restrict__ A, int lda, int K,
    const u16* __restrict__ B, size_t bstride,
    u16* __restrict__ O, size_t ostride,
    const float* __restrict__ bias2,
    const float* __restrict__ scale, const float* __restrict__ shift,
    const float* __restrict__ xres)
{
#if USE_TDM
  __shared__ __align__(16) u16 Bs[32 * 132];             // row-major [k][col], 2-dword row pad via TDM
#else
  __shared__ __align__(16) u16 Bs[128 * 40];             // transposed [col][k], +8 pad
#endif
  const int n   = blockIdx.z;
  const int n0  = blockIdx.x * 128;
  const int m0b = blockIdx.y * 128;
  const int lane = threadIdx.x & 31, wid = threadIdx.x >> 5;
  const int waveM = wid >> 1, waveN = wid & 1;
  const int m0w = m0b + waveM * 32;
  const int n0w = waveN * 64;
  const u16* Bn = B + (size_t)n * bstride;
  u16* On = O + (size_t)n * ostride;

#if USE_TDM
  // LDS-aperture generic address carries the LDS byte offset in addr[31:0] (ISA 10.2)
  const u32 lds_base = (u32)(uintptr_t)(void*)Bs;
#endif

  FragC acc[2][4];
#pragma unroll
  for (int i = 0; i < 2; ++i)
#pragma unroll
    for (int j = 0; j < 4; ++j)
#pragma unroll
      for (int r = 0; r < 8; ++r) acc[i][j].f[r] = 0.f;

  for (int k0 = 0; k0 < K; k0 += 32) {
    __syncthreads();
#if USE_TDM
    if (wid == 0) {
      const u16* gsrc = Bn + (size_t)k0 * LQ + n0;
      const uint64_t ga = (uint64_t)(uintptr_t)gsrc;
      // D# group0: count=1 | lds_addr | global_addr[56:0] | type=2
      v4u g0 = { 1u, lds_base, (u32)ga, (u32)(ga >> 32) | 0x80000000u };
      // D# group1: data_size=2B, pad_enable, pad_interval=64dw, pad_amount=2dw,
      //            tensor_dim0=LQ, tensor_dim1=K, tile_dim0=128, tile_dim1=32, dim0_stride=LQ
      v8i g1 = { (int)0x03510000u,
                 (int)(((u32)LQ & 0xFFFFu) << 16),
                 (int)(((u32)LQ >> 16) | (((u32)K & 0xFFFFu) << 16)),
                 (int)(((u32)K >> 16) | (128u << 16)),
                 32, LQ, 0, 0 };
      v4i zz = { 0, 0, 0, 0 };
      v8i z8 = { 0, 0, 0, 0, 0, 0, 0, 0 };
      __builtin_amdgcn_tensor_load_to_lds(g0, g1, zz, zz, z8, 0);
      __builtin_amdgcn_s_wait_tensorcnt(0);
    }
    __syncthreads();
#else
    for (int i = threadIdx.x; i < 512; i += 256) {       // 32 rows x 16 chunks of 8 cols
      int k = i >> 4, ch = i & 15;
      U16B d = *reinterpret_cast<const U16B*>(Bn + (size_t)(k0 + k) * LQ + n0 + ch * 8);
      if (k0 + 32 < K)
        __builtin_prefetch(Bn + (size_t)(k0 + 32 + k) * LQ + n0 + ch * 8, 0, 3);
      union { U16B q; u16 u[8]; } w; w.q = d;
#pragma unroll
      for (int j = 0; j < 8; ++j) Bs[(ch * 8 + j) * 40 + k] = w.u[j];
    }
    __syncthreads();
#endif

    FragAB a[2];
#pragma unroll
    for (int fa = 0; fa < 2; ++fa)
      load_a(a[fa], A, m0w + fa * 16 + (lane & 15), lda, k0, lane);

    FragAB b[4];
#pragma unroll
    for (int ct = 0; ct < 4; ++ct) {
      const int col = n0w + ct * 16 + (lane & 15);
      const int kb = (lane >> 4) * 16;
#if USE_TDM
#pragma unroll
      for (int j = 0; j < 16; ++j) b[ct].u[j] = Bs[(kb + j) * 132 + col];
#else
      const U16B* p = reinterpret_cast<const U16B*>(&Bs[col * 40 + kb]);
      b[ct].q[0] = p[0];
      b[ct].q[1] = p[1];
#endif
    }
#pragma unroll
    for (int fa = 0; fa < 2; ++fa)
#pragma unroll
      for (int ct = 0; ct < 4; ++ct)
        acc[fa][ct].v = wmma_bf16(a[fa].v, b[ct].v, acc[fa][ct].v);
  }

#pragma unroll
  for (int fa = 0; fa < 2; ++fa)
#pragma unroll
    for (int ct = 0; ct < 4; ++ct) {
      const int colL = n0 + n0w + ct * 16 + (lane & 15);
      const int vv = colL % VP;
      const int t  = colL / VP;
#pragma unroll
      for (int r = 0; r < 8; ++r) {
        const int row = m0w + fa * 16 + r + 8 * (lane >> 4);
        const float va = acc[fa][ct].f[r];
        u16 o = 0;
        if (MODE == 0) {
          if (vv < VV) o = f2bf(va + bias2[row * VP + vv]);
        } else {
          if (vv < VV) {
            float xv = xres[(((size_t)n * CC + row) * TT + t) * VV + vv];
            float z = va * scale[row] + shift[row];
            o = f2bf(lrelu(xv + z));
          }
        }
        On[(size_t)row * LQ + colL] = o;
      }
    }
}

// ---------------- attention accumulation: att[u,v] += sum_{i,t} q*k ----------------
__global__ __launch_bounds__(256) void k_att_acc(const u16* __restrict__ qk, float* __restrict__ attacc) {
  __shared__ __align__(16) u16 Qs[80 * 40];              // transposed [u][i], +8 pad
  __shared__ __align__(16) u16 Ks[80 * 40];              // transposed [v][i], +8 pad
  const int ns = blockIdx.y;
  const int n = ns >> 1, s = ns & 1;
  const int lane = threadIdx.x & 31, wid = threadIdx.x >> 5;
  const u16* qn = qk + (size_t)n * PLANE + (size_t)(s * NI) * LQ;
  const u16* kn = qk + (size_t)n * PLANE + (size_t)((SS + s) * NI) * LQ;
  const int c0 = blockIdx.x * 50;                        // 800 K-chunks split 16 ways

  FragC acc[4];                                          // tiles wid, wid+8, wid+16, wid+24
#pragma unroll
  for (int j = 0; j < 4; ++j)
#pragma unroll
    for (int r = 0; r < 8; ++r) acc[j].f[r] = 0.f;

  for (int kc = 0; kc < 50; ++kc) {
    const int kk = (c0 + kc) * 32;                       // k = t*64 + i
    const int t  = kk >> 6;
    const int ib = kk & 63;
    __syncthreads();
    for (int i = threadIdx.x; i < 640; i += 256) {       // q: 32 rows x 10 chunks, k: same
      const int which = (i >= 320) ? 1 : 0;
      const int idx = i - which * 320;
      const int krow = idx / 10, ch = idx % 10;
      const u16* src = (which ? kn : qn) + (size_t)(ib + krow) * LQ + t * VP + ch * 8;
      U16B d = *reinterpret_cast<const U16B*>(src);
      union { U16B q; u16 u[8]; } wv; wv.q = d;
      u16* dst = which ? Ks : Qs;
#pragma unroll
      for (int j = 0; j < 8; ++j) dst[(ch * 8 + j) * 40 + krow] = wv.u[j];
    }
    __syncthreads();

#pragma unroll
    for (int j = 0; j < 4; ++j) {
      const int tile = wid + 8 * j;                      // wave-uniform
      if (tile < 25) {
        const int ut = tile / 5, vt = tile % 5;
        FragAB a, b;
        load_a(a, Qs, ut * 16 + (lane & 15), 40, 0, lane);
        const int col = vt * 16 + (lane & 15);
        const U16B* p = reinterpret_cast<const U16B*>(&Ks[col * 40 + (lane >> 4) * 16]);
        b.q[0] = p[0];
        b.q[1] = p[1];
        acc[j].v = wmma_bf16(a.v, b.v, acc[j].v);
      }
    }
  }

  float* dst = attacc + (size_t)ns * 6400;
#pragma unroll
  for (int j = 0; j < 4; ++j) {
    const int tile = wid + 8 * j;
    if (tile < 25) {
      const int ut = tile / 5, vt = tile % 5;
#pragma unroll
      for (int r = 0; r < 8; ++r) {
        int row = ut * 16 + r + 8 * (lane >> 4);
        int col = vt * 16 + (lane & 15);
        atomicAdd(&dst[row * VP + col], acc[j].f[r]);
      }
    }
  }
}

__global__ void k_att_fin(const float* __restrict__ attacc, const float* __restrict__ alphas,
                          const float* __restrict__ att0, u16* __restrict__ attbf) {
  int idx = blockIdx.x * 256 + threadIdx.x;              // 8*96*80
  int ns = idx / (96 * 80);
  int r  = idx % (96 * 80);
  int u = r / 80, v = r % 80;
  int s = ns & 1;
  float val = 0.f;
  if (u < VV && v < VV) {
    float a = attacc[(size_t)ns * 6400 + u * VP + v] * (1.f / 25600.f);
    val = tanhf(a) * alphas[s] + att0[((size_t)s * VV + u) * VV + v];
  }
  attbf[idx] = f2bf(val);                                // [ns][96][80], zero padded
}

// ---------------- apply attention: h[s,c,t,v] = sum_u x[c,t,u] * att[s,u,v] ----------------
__global__ __launch_bounds__(256) void k_apply_att(const u16* __restrict__ xbf,
                                                   const u16* __restrict__ attbf,
                                                   u16* __restrict__ h) {
  __shared__ __align__(16) u16 AattT[80 * 104];          // transposed [v][u], +8 pad
  const int ns = blockIdx.y;
  const int n = ns >> 1;
  const int lane = threadIdx.x & 31, wid = threadIdx.x >> 5;
  for (int i = threadIdx.x; i < 7680; i += 256) {        // [u][v] -> [v][u]
    int u = i / 80, v = i % 80;
    AattT[v * 104 + u] = attbf[(size_t)ns * 7680 + i];
  }
  __syncthreads();

  const u16* xb = xbf + (size_t)n * PLANE;
  const int mrowbase = blockIdx.x * 128 + wid * 16;
  const int m = mrowbase + (lane & 15);

  FragC acc[5];
#pragma unroll
  for (int ct = 0; ct < 5; ++ct)
#pragma unroll
    for (int r = 0; r < 8; ++r) acc[ct].f[r] = 0.f;

  for (int k0 = 0; k0 < 96; k0 += 32) {
    FragAB a;
    if (k0 < 64) {
      load_a(a, xb, m, VP, k0, lane);
    } else {                                             // K tail: rows 64..79 valid, rest zero
      int h2 = lane >> 4;
      if (h2 == 0) {
        a.q[0] = *reinterpret_cast<const U16B*>(xb + (size_t)m * VP + 64);
        a.q[1] = *reinterpret_cast<const U16B*>(xb + (size_t)m * VP + 72);
      } else {
        a.q[0].d[0] = a.q[0].d[1] = a.q[0].d[2] = a.q[0].d[3] = 0;
        a.q[1].d[0] = a.q[1].d[1] = a.q[1].d[2] = a.q[1].d[3] = 0;
      }
    }
#pragma unroll
    for (int ct = 0; ct < 5; ++ct) {
      FragAB b;
      const int v = ct * 16 + (lane & 15);
      const int kb = k0 + (lane >> 4) * 16;
      const U16B* p = reinterpret_cast<const U16B*>(&AattT[v * 104 + kb]);
      b.q[0] = p[0];
      b.q[1] = p[1];
      acc[ct].v = wmma_bf16(a.v, b.v, acc[ct].v);
    }
  }

  u16* hn = h + (size_t)ns * PLANE;
#pragma unroll
  for (int ct = 0; ct < 5; ++ct)
#pragma unroll
    for (int r = 0; r < 8; ++r) {
      int row = mrowbase + r + 8 * (lane >> 4);
      int col = ct * 16 + (lane & 15);
      hn[(size_t)row * VP + col] = f2bf(acc[ct].f[r]);
    }
}

// ---------------- temporal conv (K=768 GEMM with t-shifted B) + BN + residual ----------------
__global__ __launch_bounds__(256) void k_gemm_conv(
    const u16* __restrict__ A,                           // w_t_bf [256][768] = [o][dt*256+c]
    const u16* __restrict__ y2,                          // bf16 [n][256][LQ]
    const float* __restrict__ scale, const float* __restrict__ shift,
    float* __restrict__ out)
{
  __shared__ __align__(16) u16 Bs[128 * 40];
  const int n   = blockIdx.z;
  const int n0  = blockIdx.x * 128;
  const int m0b = blockIdx.y * 128;
  const int lane = threadIdx.x & 31, wid = threadIdx.x >> 5;
  const int waveM = wid >> 1, waveN = wid & 1;
  const int m0w = m0b + waveM * 32;
  const int n0w = waveN * 64;
  const u16* Bn = y2 + (size_t)n * PLANE;

  FragC acc[2][4];
#pragma unroll
  for (int i = 0; i < 2; ++i)
#pragma unroll
    for (int j = 0; j < 4; ++j)
#pragma unroll
      for (int r = 0; r < 8; ++r) acc[i][j].f[r] = 0.f;

  for (int dt = 0; dt < 3; ++dt) {
    const int shiftL = (dt - 1) * VP;
    for (int kc = 0; kc < 8; ++kc) {
      __syncthreads();
      for (int i = threadIdx.x; i < 512; i += 256) {
        int k = i >> 4, ch = i & 15;
        long Lp = (long)n0 + ch * 8 + shiftL;            // 8-col chunks never straddle t
        U16B d;
        if (Lp >= 0 && Lp < LQ)
          d = *reinterpret_cast<const U16B*>(Bn + (size_t)(kc * 32 + k) * LQ + Lp);
        else { d.d[0] = d.d[1] = d.d[2] = d.d[3] = 0; }
        if (kc < 7 && Lp >= 0 && Lp < LQ)
          __builtin_prefetch(Bn + (size_t)(kc * 32 + k + 32) * LQ + Lp, 0, 3);
        union { U16B q; u16 u[8]; } w; w.q = d;
#pragma unroll
        for (int j = 0; j < 8; ++j) Bs[(ch * 8 + j) * 40 + k] = w.u[j];
      }
      __syncthreads();

      const int kA = dt * 256 + kc * 32;
      FragAB a[2];
#pragma unroll
      for (int fa = 0; fa < 2; ++fa)
        load_a(a[fa], A, m0w + fa * 16 + (lane & 15), 768, kA, lane);

      FragAB b[4];
#pragma unroll
      for (int ct = 0; ct < 4; ++ct) {
        int col = n0w + ct * 16 + (lane & 15);
        int kb = (lane >> 4) * 16;
        const U16B* p = reinterpret_cast<const U16B*>(&Bs[col * 40 + kb]);
        b[ct].q[0] = p[0];
        b[ct].q[1] = p[1];
      }
#pragma unroll
      for (int fa = 0; fa < 2; ++fa)
#pragma unroll
        for (int ct = 0; ct < 4; ++ct)
          acc[fa][ct].v = wmma_bf16(a[fa].v, b[ct].v, acc[fa][ct].v);
    }
  }

#pragma unroll
  for (int fa = 0; fa < 2; ++fa)
#pragma unroll
    for (int ct = 0; ct < 4; ++ct) {
      const int colL = n0 + n0w + ct * 16 + (lane & 15);
      const int vv = colL % VP;
      const int t  = colL / VP;
      if (vv < VV) {
#pragma unroll
        for (int r = 0; r < 8; ++r) {
          const int row = m0w + fa * 16 + r + 8 * (lane >> 4);
          float z = acc[fa][ct].f[r] * scale[row] + shift[row];
          float y2v = bf2f(Bn[(size_t)row * LQ + colL]);
          out[(((size_t)n * CC + row) * TT + t) * VV + vv] = lrelu(y2v + z);
        }
      }
    }
}

// ---------------- host side ----------------
extern "C" void kernel_launch(void* const* d_in, const int* in_sizes, int n_in,
                              void* d_out, int out_size, void* d_ws, size_t ws_size,
                              hipStream_t stream) {
  (void)in_sizes; (void)n_in; (void)out_size; (void)ws_size;

  const float* x      = (const float*)d_in[0];
  const float* w_in   = (const float*)d_in[1];
  const float* b_in   = (const float*)d_in[2];
  const float* alphas = (const float*)d_in[3];
  const float* att0   = (const float*)d_in[4];
  const float* w_out  = (const float*)d_in[5];
  const float* b_out  = (const float*)d_in[6];
  const float* g1 = (const float*)d_in[7],  *bt1 = (const float*)d_in[8];
  const float* m1 = (const float*)d_in[9],  *v1  = (const float*)d_in[10];
  const float* w_ff = (const float*)d_in[11], *b_ff = (const float*)d_in[12];
  const float* g2 = (const float*)d_in[13], *bt2 = (const float*)d_in[14];
  const float* m2 = (const float*)d_in[15], *v2  = (const float*)d_in[16];
  const float* w_t = (const float*)d_in[17], *b_t = (const float*)d_in[18];
  const float* g3 = (const float*)d_in[19], *bt3 = (const float*)d_in[20];
  const float* m3 = (const float*)d_in[21], *v3  = (const float*)d_in[22];
  float* outp = (float*)d_out;

  // workspace layout (bytes); overlays: qk->y1, x_bf->y2
  constexpr size_t SZ_XBF = (size_t)NB * PLANE * 2;      //  65,536,000 (x_bf, later y2)
  constexpr size_t SZ_QK  = SZ_XBF;                      //  (qk, later y1)
  constexpr size_t SZ_H   = (size_t)NB * 2 * PLANE * 2;  // 131,072,000
  constexpr size_t OFF_XBF  = 0;
  constexpr size_t OFF_QK   = OFF_XBF + SZ_XBF;
  constexpr size_t OFF_H    = OFF_QK + SZ_QK;
  constexpr size_t OFF_WIN  = OFF_H + SZ_H;
  constexpr size_t OFF_WOUT = OFF_WIN  + 256 * 256 * 2;
  constexpr size_t OFF_WFF  = OFF_WOUT + 256 * 512 * 2;
  constexpr size_t OFF_WT   = OFF_WFF  + 256 * 256 * 2;
  constexpr size_t OFF_B2   = OFF_WT   + 256 * 768 * 2;
  constexpr size_t OFF_AAC  = OFF_B2   + 256 * 80 * 4;
  constexpr size_t OFF_ABF  = OFF_AAC  + 8 * 6400 * 4;
  constexpr size_t OFF_BNP  = OFF_ABF  + 8 * 96 * 80 * 2;

  char* w = (char*)d_ws;
  u16*   xbf   = (u16*)(w + OFF_XBF);
  u16*   qkb   = (u16*)(w + OFF_QK);
  u16*   hb    = (u16*)(w + OFF_H);
  u16*   winb  = (u16*)(w + OFF_WIN);
  u16*   woutb = (u16*)(w + OFF_WOUT);
  u16*   wffb  = (u16*)(w + OFF_WFF);
  u16*   wtb   = (u16*)(w + OFF_WT);
  float* bias2 = (float*)(w + OFF_B2);
  float* aacc  = (float*)(w + OFF_AAC);
  u16*   abf   = (u16*)(w + OFF_ABF);
  float* bnp   = (float*)(w + OFF_BNP);
  u16*   y1b   = qkb;   // overlay
  u16*   y2b   = xbf;   // overlay

  // prep
  k_prep_x<<<128000, 256, 0, stream>>>(x, xbf);
  k_prep_w<<<1792, 256, 0, stream>>>(w_in, w_out, w_ff, w_t, winb, woutb, wffb, wtb);
  k_prep_bias2<<<80, 256, 0, stream>>>(w_in, b_in, bias2);
  k_prep_bn<<<1, 256, 0, stream>>>(b_out, g1, bt1, m1, v1, b_ff, g2, bt2, m2, v2,
                                   b_t, g3, bt3, m3, v3, bnp);
  k_zero_att<<<200, 256, 0, stream>>>(aacc);

  // stage A: qk = W_in * x + bias2(pe)
  k_gemm_main<0><<<dim3(250, 2, 4), 256, 0, stream>>>(
      winb, 256, 256, xbf, PLANE, qkb, PLANE, bias2, nullptr, nullptr, nullptr);

  // stage B: attention
  k_att_acc<<<dim3(16, 8), 256, 0, stream>>>(qkb, aacc);
  k_att_fin<<<240, 256, 0, stream>>>(aacc, alphas, att0, abf);

  // stage C: h = x @ att
  k_apply_att<<<dim3(800, 8), 256, 0, stream>>>(xbf, abf, hb);

  // stage D: y1 = lrelu(x + BN(W_out * h))
  k_gemm_main<1><<<dim3(250, 2, 4), 256, 0, stream>>>(
      woutb, 512, 512, hb, 2 * PLANE, y1b, PLANE, nullptr, bnp + 0, bnp + 256, x);

  // stage E: y2 = lrelu(x + BN(W_ff * y1))
  k_gemm_main<1><<<dim3(250, 2, 4), 256, 0, stream>>>(
      wffb, 256, 256, y1b, PLANE, y2b, PLANE, nullptr, bnp + 512, bnp + 768, x);

  // stage F: out = lrelu(y2 + BN(conv_t(y2)))
  k_gemm_conv<<<dim3(250, 2, 4), 256, 0, stream>>>(wtb, y2b, bnp + 1024, bnp + 1280, outp);
}
